// ModuleWithControlFlowLoopScan_27230092657008
// MI455X (gfx1250) — matrix operand order repaired and checked by reading, not compiled, not run on hardware
//
#include <hip/hip_runtime.h>

// -------- CDNA5 wave32 WMMA pairwise-distance GEMM --------
// d[m][n] = sqrt(max(||x_m||^2 + ||y_n||^2 - 2 <x_m, y_n>, 0))
// Cross term via v_wmma_f32_16x16x32_bf16; norms in exact f32.
// Inputs converted to bf16 once in a fused norm+convert prologue; the GEMM
// streams bf16 tiles into LDS with GLOBAL_LOAD_ASYNC_TO_LDS_B128 (ASYNCcnt),
// then feeds WMMA from LDS via ds_load_b128.

typedef __bf16 bf16x8  __attribute__((ext_vector_type(8)));
typedef __bf16 bf16x16 __attribute__((ext_vector_type(16)));
typedef float  f32x8   __attribute__((ext_vector_type(8)));

#define NX 4096
#define NY 16384
#define KD 256
#define BM 128
#define BN 128
#define KT 64
#define PITCH 72  // KT + 8 bf16 pad: 144B row pitch, 16B aligned, bank-conflict free

union AFrag { bf16x16 v16; bf16x8 v8[2]; };

// Pack two f32 -> two bf16 (round-half-up) in 3 VALU ops: 2 adds + v_perm_b32.
__device__ __forceinline__ unsigned pack_bf16x2(float a, float b) {
  unsigned ua = __builtin_bit_cast(unsigned, a) + 0x8000u;
  unsigned ub = __builtin_bit_cast(unsigned, b) + 0x8000u;
  return __builtin_amdgcn_perm(ub, ua, 0x07060302u);  // { ub.hi, ua.hi }
}

// Async copy of 16 bytes global -> LDS (GV mode), tracked with ASYNCcnt.
__device__ __forceinline__ void async_b128(unsigned lds_addr, const void* gptr) {
  asm volatile("global_load_async_to_lds_b128 %0, %1, off"
               :: "v"(lds_addr), "v"((unsigned long long)gptr)
               : "memory");
}
__device__ __forceinline__ void wait_async0() {
  asm volatile("s_wait_asynccnt 0" ::: "memory");
}

// ---------------- Prologue: fused row-norm + f32->bf16 convert ----------------
__global__ __launch_bounds__(256) void cvt_norm(const float* __restrict__ v,
                                                unsigned short* __restrict__ vb,
                                                float* __restrict__ nrm) {
  int row  = blockIdx.x * 8 + (threadIdx.x >> 5);
  int lane = threadIdx.x & 31;
  const float4* p = (const float4*)(v + (size_t)row * KD) + lane * 2;
  float4 f0 = p[0];
  float4 f1 = p[1];
  float s = f0.x * f0.x + f0.y * f0.y + f0.z * f0.z + f0.w * f0.w +
            f1.x * f1.x + f1.y * f1.y + f1.z * f1.z + f1.w * f1.w;
  uint4 pk;
  pk.x = pack_bf16x2(f0.x, f0.y);
  pk.y = pack_bf16x2(f0.z, f0.w);
  pk.z = pack_bf16x2(f1.x, f1.y);
  pk.w = pack_bf16x2(f1.z, f1.w);
  *(uint4*)(vb + (size_t)row * KD + lane * 8) = pk;
#pragma unroll
  for (int off = 16; off > 0; off >>= 1) s += __shfl_xor(s, off, 32);
  if (lane == 0) nrm[row] = s;
}

// Fallback prologue (norms only) when workspace can't hold bf16 copies.
__global__ __launch_bounds__(256) void row_sqnorm(const float* __restrict__ v,
                                                  float* __restrict__ out) {
  int row  = blockIdx.x * 8 + (threadIdx.x >> 5);
  int lane = threadIdx.x & 31;
  const float4* p = (const float4*)(v + (size_t)row * KD) + lane * 2;
  float s = 0.f;
#pragma unroll
  for (int i = 0; i < 2; ++i) {
    float4 f = p[i];
    s += f.x * f.x + f.y * f.y + f.z * f.z + f.w * f.w;
  }
#pragma unroll
  for (int off = 16; off > 0; off >>= 1) s += __shfl_xor(s, off, 32);
  if (lane == 0) out[row] = s;
}

// ---------------- Shared GEMM core pieces ----------------
struct WaveCtx {
  int lane, waveM, waveN, m0, n0;
};

__device__ __forceinline__ void wmma_step(const unsigned short* sA,
                                          const unsigned short* sB,
                                          const WaveCtx& w, f32x8 acc[2][4]) {
#pragma unroll
  for (int kk = 0; kk < KT; kk += 32) {
    AFrag a[2], b[4];
    // A 16x32 bf16 layout: lanes 0-15 K={0..7,16..23}, lanes 16-31 K={8..15,24..31}
    const int ka = kk + ((w.lane & 16) ? 8 : 0);
    // B 32x16 bf16 layout: lanes 0-15 K={0..7,16..23}, lanes 16-31 K={8..15,24..31}
    const int kb = kk + ((w.lane & 16) ? 8 : 0);
#pragma unroll
    for (int mt = 0; mt < 2; ++mt) {
      int r = w.waveM * 32 + mt * 16 + (w.lane & 15);
      a[mt].v8[0] = *(const bf16x8*)&sA[r * PITCH + ka];
      a[mt].v8[1] = *(const bf16x8*)&sA[r * PITCH + ka + 16];
    }
#pragma unroll
    for (int nt = 0; nt < 4; ++nt) {
      int c = w.waveN * 64 + nt * 16 + (w.lane & 15);
      b[nt].v8[0] = *(const bf16x8*)&sB[c * PITCH + kb];
      b[nt].v8[1] = *(const bf16x8*)&sB[c * PITCH + kb + 16];
    }
#pragma unroll
    for (int mt = 0; mt < 2; ++mt)
#pragma unroll
      for (int nt = 0; nt < 4; ++nt)
        acc[mt][nt] = __builtin_amdgcn_wmma_f32_16x16x32_bf16(
            false, a[mt].v16, false, b[nt].v16, (short)0, acc[mt][nt],
            false, false);
  }
}

__device__ __forceinline__ void epilogue(const float* sx2, const float* sy2,
                                         const WaveCtx& w, f32x8 acc[2][4],
                                         float* __restrict__ out) {
  const int mhalf = (w.lane & 16) ? 8 : 0;
#pragma unroll
  for (int mt = 0; mt < 2; ++mt) {
#pragma unroll
    for (int nt = 0; nt < 4; ++nt) {
      int    nl   = w.waveN * 64 + nt * 16 + (w.lane & 15);
      float  yn   = sy2[nl];
      size_t ncol = (size_t)(w.n0 + nl);
#pragma unroll
      for (int r = 0; r < 8; ++r) {
        int   ml = w.waveM * 32 + mt * 16 + mhalf + r;
        float d2 = sx2[ml] + yn - 2.0f * acc[mt][nt][r];
        out[(size_t)(w.m0 + ml) * NY + ncol] = sqrtf(fmaxf(d2, 0.f));
      }
    }
  }
}

// ---------------- Main GEMM (pre-converted bf16, async LDS staging) ----------------
__global__ __launch_bounds__(256) void pdist_wmma_bf16(
    const unsigned short* __restrict__ Xb, const unsigned short* __restrict__ Yb,
    const float* __restrict__ x2, const float* __restrict__ y2,
    float* __restrict__ out) {
  __shared__ unsigned short sA[BM * PITCH];
  __shared__ unsigned short sB[BN * PITCH];
  __shared__ float sx2[BM];
  __shared__ float sy2[BN];

  const int tid = threadIdx.x;
  WaveCtx w;
  w.lane  = tid & 31;
  w.waveM = (tid >> 5) & 3;   // 4 waves over M: 32 rows each
  w.waveN = tid >> 7;         // 2 waves over N: 64 cols each
  w.m0 = blockIdx.y * BM;
  w.n0 = blockIdx.x * BN;

  if (tid < BM) sx2[tid] = x2[w.m0 + tid];
  else          sy2[tid - BM] = y2[w.n0 + (tid - BM)];

  f32x8 acc[2][4];
#pragma unroll
  for (int i = 0; i < 2; ++i)
#pragma unroll
    for (int j = 0; j < 4; ++j)
      acc[i][j] = (f32x8){0.f, 0.f, 0.f, 0.f, 0.f, 0.f, 0.f, 0.f};

  for (int k0 = 0; k0 < KD; k0 += KT) {
    __syncthreads();
    // Stage 128x64 bf16 tiles straight into LDS via async b128 copies.
#pragma unroll
    for (int it = 0; it < 4; ++it) {
      int i   = tid + it * 256;    // 0..1023 : 128 rows x 8 8-elem chunks
      int row = i >> 3;
      int c8  = (i & 7) << 3;      // bf16 column within K-tile
      unsigned la = (unsigned)(unsigned long long)&sA[row * PITCH + c8];
      unsigned lb = (unsigned)(unsigned long long)&sB[row * PITCH + c8];
      async_b128(la, Xb + (size_t)(w.m0 + row) * KD + k0 + c8);
      async_b128(lb, Yb + (size_t)(w.n0 + row) * KD + k0 + c8);
    }
    wait_async0();          // all LDS writes from this wave complete
    __syncthreads();        // ... and visible to every wave
    wmma_step(sA, sB, w, acc);
  }

  epilogue(sx2, sy2, w, acc, out);
}

// ---------------- Fallback GEMM (in-kernel f32->bf16 conversion) ----------------
__global__ __launch_bounds__(256) void pdist_wmma_f32(
    const float* __restrict__ X, const float* __restrict__ Y,
    const float* __restrict__ x2, const float* __restrict__ y2,
    float* __restrict__ out) {
  __shared__ unsigned short sA[BM * PITCH];
  __shared__ unsigned short sB[BN * PITCH];
  __shared__ float sx2[BM];
  __shared__ float sy2[BN];

  const int tid = threadIdx.x;
  WaveCtx w;
  w.lane  = tid & 31;
  w.waveM = (tid >> 5) & 3;
  w.waveN = tid >> 7;
  w.m0 = blockIdx.y * BM;
  w.n0 = blockIdx.x * BN;

  if (tid < BM) sx2[tid] = x2[w.m0 + tid];
  else          sy2[tid - BM] = y2[w.n0 + (tid - BM)];

  f32x8 acc[2][4];
#pragma unroll
  for (int i = 0; i < 2; ++i)
#pragma unroll
    for (int j = 0; j < 4; ++j)
      acc[i][j] = (f32x8){0.f, 0.f, 0.f, 0.f, 0.f, 0.f, 0.f, 0.f};

  for (int k0 = 0; k0 < KD; k0 += KT) {
    __syncthreads();
#pragma unroll
    for (int it = 0; it < 8; ++it) {
      int i   = tid + it * 256;    // 0..2047 : 128 rows x 16 float4
      int row = i >> 4;
      int c4  = (i & 15) << 2;
      float4 fa = *(const float4*)(X + (size_t)(w.m0 + row) * KD + k0 + c4);
      float4 fb = *(const float4*)(Y + (size_t)(w.n0 + row) * KD + k0 + c4);
      uint2 pa = make_uint2(pack_bf16x2(fa.x, fa.y), pack_bf16x2(fa.z, fa.w));
      uint2 pb = make_uint2(pack_bf16x2(fb.x, fb.y), pack_bf16x2(fb.z, fb.w));
      *(uint2*)&sA[row * PITCH + c4] = pa;
      *(uint2*)&sB[row * PITCH + c4] = pb;
    }
    __syncthreads();
    wmma_step(sA, sB, w, acc);
  }

  epilogue(sx2, sy2, w, acc, out);
}

extern "C" void kernel_launch(void* const* d_in, const int* in_sizes, int n_in,
                              void* d_out, int out_size, void* d_ws, size_t ws_size,
                              hipStream_t stream) {
  (void)in_sizes; (void)n_in; (void)out_size;
  const float* X = (const float*)d_in[0];   // [4096, 256]
  const float* Y = (const float*)d_in[1];   // [16384, 256]
  float* out = (float*)d_out;               // [4096, 16384]

  // Workspace layout: x2[NX] f32 | y2[NY] f32 | Xb[NX*KD] bf16 | Yb[NY*KD] bf16
  float* x2 = (float*)d_ws;
  float* y2 = x2 + NX;
  unsigned short* Xb = (unsigned short*)(y2 + NY);
  unsigned short* Yb = Xb + (size_t)NX * KD;
  const size_t need = (size_t)(NX + NY) * sizeof(float) +
                      (size_t)(NX + NY) * KD * sizeof(unsigned short);

  dim3 grid(NY / BN, NX / BM);              // 128 x 32 blocks

  if (ws_size >= need) {
    cvt_norm<<<NX / 8, 256, 0, stream>>>(X, Xb, x2);
    cvt_norm<<<NY / 8, 256, 0, stream>>>(Y, Yb, y2);
    pdist_wmma_bf16<<<grid, 256, 0, stream>>>(Xb, Yb, x2, y2, out);
  } else {
    row_sqnorm<<<NX / 8, 256, 0, stream>>>(X, x2);
    row_sqnorm<<<NY / 8, 256, 0, stream>>>(Y, y2);
    pdist_wmma_f32<<<grid, 256, 0, stream>>>(X, Y, x2, y2, out);
  }
}
